// EdgeRemovalLayer_82162724372845
// MI455X (gfx1250) — matrix operand order repaired and checked by reading, not compiled, not run on hardware
//
#include <hip/hip_runtime.h>
#include <hip/hip_bf16.h>
#include <math.h>

// ---- shapes from the reference ----
#define D_NODE 64
#define D_EDGE 32
#define D_IN   160   // 2*D_NODE + D_EDGE
#define HID1   64
#define HID2   32
#define WAVES  4
#define TPB    (WAVES * 32)

typedef __attribute__((ext_vector_type(16))) _Float16 v16h;
typedef __attribute__((ext_vector_type(8)))  _Float16 v8h;
typedef __attribute__((ext_vector_type(8)))  float    v8f;

__device__ __forceinline__ v16h pack16(v8h lo, v8h hi) {
    v16h r;
#pragma unroll
    for (int i = 0; i < 8; ++i) { r[i] = lo[i]; r[i + 8] = hi[i]; }
    return r;
}

__global__ __launch_bounds__(TPB) void edge_mlp_wmma_kernel(
    const float* __restrict__ nf,          // [N_NODES, 64]
    const long long* __restrict__ eidx,    // [2, E] int64
    const float* __restrict__ ef,          // [E, 32]
    const float* __restrict__ W1, const float* __restrict__ B1,   // [64,160],[64]
    const float* __restrict__ W2, const float* __restrict__ B2,   // [32,64],[32]
    const float* __restrict__ W3, const float* __restrict__ B3,   // [1,32],[1]
    float* __restrict__ out,               // [scores(E) | mask(E)]
    int E)
{
    // ---- LDS: weights (f16) + per-wave staging ----
    __shared__ __align__(16) _Float16 w1s[HID1 * D_IN];            // 20480 B
    __shared__ float b1s[HID1];
    __shared__ __align__(16) _Float16 w2s[HID2 * HID1];            // 4096 B
    __shared__ float b2s[HID2];
    __shared__ float w3s[HID2];
    __shared__ float b3s[1];
    __shared__ __align__(16) _Float16 comb[WAVES][16 * D_IN];      // 4*5120 B
    __shared__ __align__(16) _Float16 x1s [WAVES][16 * HID1];      // 4*2048 B
    __shared__ float                  x2s [WAVES][16 * HID2];      // 4*2048 B

    const int tid  = threadIdx.x;
    const int wv   = tid >> 5;
    const int lane = tid & 31;
    const int half = lane >> 4;   // which 16-lane half of the wave
    const int mrow = lane & 15;

    // ---- stage weights (f32 -> f16 for WMMA operands) ----
    for (int i = tid; i < HID1 * D_IN; i += TPB) w1s[i] = (_Float16)W1[i];
    for (int i = tid; i < HID2 * HID1; i += TPB) w2s[i] = (_Float16)W2[i];
    if (tid < HID1) b1s[tid] = B1[tid];
    if (tid < HID2) { b2s[tid] = B2[tid]; w3s[tid] = W3[tid]; }
    if (tid == 0) b3s[0] = B3[0];

    // ---- stage combined = [src | tgt | edge] rows as f16 (coalesced) ----
    const long long e0 = (long long)blockIdx.x * (WAVES * 16) + (long long)wv * 16;
    for (int m = 0; m < 16; ++m) {
        long long e = e0 + m; if (e >= E) e = E - 1;           // clamp: safe gather
        const long long s = eidx[e];
        const long long t = eidx[(long long)E + e];
#pragma unroll
        for (int j = 0; j < 5; ++j) {
            const int k = lane + 32 * j;
            float v;
            if (k < D_NODE)            v = nf[s * D_NODE + k];
            else if (k < 2 * D_NODE)   v = nf[t * D_NODE + (k - D_NODE)];
            else                       v = ef[e * D_EDGE + (k - 2 * D_NODE)];
            comb[wv][m * D_IN + k] = (_Float16)v;
        }
    }
    __syncthreads();

    // ================= Layer 1: [16x160] x [160x64] =================
    const v8f vzero = {0.f, 0.f, 0.f, 0.f, 0.f, 0.f, 0.f, 0.f};
    v8f acc1[4];
#pragma unroll
    for (int nt = 0; nt < 4; ++nt) acc1[nt] = vzero;

#pragma unroll
    for (int kt = 0; kt < 5; ++kt) {
        const int kb = kt * 32;
        // A fragment: lane half covers K = kb+half*8..+7 and kb+16+half*8..+7
        const v8h alo = *(const v8h*)&comb[wv][mrow * D_IN + kb + half * 8];
        const v8h ahi = *(const v8h*)&comb[wv][mrow * D_IN + kb + 16 + half * 8];
        const v16h a = pack16(alo, ahi);
#pragma unroll
        for (int nt = 0; nt < 4; ++nt) {
            const int n = nt * 16 + mrow;
            // B fragment: lane half covers K = kb+half*8..+7 and kb+16+half*8..+7
            const v8h blo = *(const v8h*)&w1s[n * D_IN + kb + half * 8];
            const v8h bhi = *(const v8h*)&w1s[n * D_IN + kb + 16 + half * 8];
            const v16h b = pack16(blo, bhi);
            acc1[nt] = __builtin_amdgcn_wmma_f32_16x16x32_f16(
                false, a, false, b, (short)0, acc1[nt], false, false);
        }
    }

    // bias + relu, write x1 (f16) in row-major for layer-2 A fragments
#pragma unroll
    for (int nt = 0; nt < 4; ++nt) {
        const int n = nt * 16 + mrow;
        const float bias = b1s[n];
#pragma unroll
        for (int r = 0; r < 8; ++r) {
            float v = acc1[nt][r] + bias;
            v = v > 0.f ? v : 0.f;
            x1s[wv][(half * 8 + r) * HID1 + n] = (_Float16)v;
        }
    }
    __syncthreads();

    // ================= Layer 2: [16x64] x [64x32] =================
    v8f acc2[2];
    acc2[0] = vzero; acc2[1] = vzero;
#pragma unroll
    for (int kt = 0; kt < 2; ++kt) {
        const int kb = kt * 32;
        const v8h alo = *(const v8h*)&x1s[wv][mrow * HID1 + kb + half * 8];
        const v8h ahi = *(const v8h*)&x1s[wv][mrow * HID1 + kb + 16 + half * 8];
        const v16h a = pack16(alo, ahi);
#pragma unroll
        for (int nt = 0; nt < 2; ++nt) {
            const int n = nt * 16 + mrow;
            const v8h blo = *(const v8h*)&w2s[n * HID1 + kb + half * 8];
            const v8h bhi = *(const v8h*)&w2s[n * HID1 + kb + 16 + half * 8];
            const v16h b = pack16(blo, bhi);
            acc2[nt] = __builtin_amdgcn_wmma_f32_16x16x32_f16(
                false, a, false, b, (short)0, acc2[nt], false, false);
        }
    }

    // bias + relu, write x2 (f32) for the tiny final layer
#pragma unroll
    for (int nt = 0; nt < 2; ++nt) {
        const int n = nt * 16 + mrow;
        const float bias = b2s[n];
#pragma unroll
        for (int r = 0; r < 8; ++r) {
            float v = acc2[nt][r] + bias;
            v = v > 0.f ? v : 0.f;
            x2s[wv][(half * 8 + r) * HID2 + n] = v;
        }
    }
    __syncthreads();

    // ================= Layer 3: [16x32] x [32x1] + sigmoid =================
    if (lane < 16) {
        const int m = lane;
        float s = b3s[0];
#pragma unroll
        for (int n = 0; n < HID2; ++n) s += x2s[wv][m * HID2 + n] * w3s[n];
        const float score = 1.0f / (1.0f + __expf(-s));
        const long long e = e0 + m;
        if (e < E) {
            out[e] = score;                                    // edge_scores
            out[(long long)E + e] = (score >= 0.5f) ? 1.f : 0.f; // keep_mask
        }
    }
}

extern "C" void kernel_launch(void* const* d_in, const int* in_sizes, int n_in,
                              void* d_out, int out_size, void* d_ws, size_t ws_size,
                              hipStream_t stream) {
    const float*     nf   = (const float*)d_in[0];
    const long long* eidx = (const long long*)d_in[1];
    const float*     ef   = (const float*)d_in[2];
    const float*     W1   = (const float*)d_in[3];
    const float*     B1   = (const float*)d_in[4];
    const float*     W2   = (const float*)d_in[5];
    const float*     B2   = (const float*)d_in[6];
    const float*     W3   = (const float*)d_in[7];
    const float*     B3   = (const float*)d_in[8];
    float* out = (float*)d_out;

    const int E = in_sizes[2] / D_EDGE;             // 1,000,000
    const int edges_per_block = WAVES * 16;         // 64
    const int blocks = (E + edges_per_block - 1) / edges_per_block;

    hipLaunchKernelGGL(edge_mlp_wmma_kernel, dim3(blocks), dim3(TPB), 0, stream,
                       nf, eidx, ef, W1, B1, W2, B2, W3, B3, out, E);
}
